// InfiniAttention_68822555951424
// MI455X (gfx1250) — compile-verified, matrix operand order checked by reading, not run on hardware
//
#include <hip/hip_runtime.h>

// Problem dims (fixed by the reference).
#define B_DIM 2
#define S_DIM 2048
#define E_DIM 1024
#define H_DIM 16
#define D_DIM 64
#define HD_DIM (H_DIM * D_DIM)

typedef __attribute__((ext_vector_type(16))) __bf16 v16bf;
typedef __attribute__((ext_vector_type(8)))  float  v8f;

union FragBF {
  v16bf v;
  unsigned int u[8];
};

// A 16x32 bf16 WMMA fragment per lane = two contiguous 16B chunks:
// elements [8*half, 8*half+8) and [16+8*half, 16+8*half+8).
__device__ __forceinline__ void load_frag(FragBF& f, const __bf16* base, int half) {
  const uint4 lo = *(const uint4*)(base + 8 * half);
  const uint4 hi = *(const uint4*)(base + 16 + 8 * half);
  f.u[0] = lo.x; f.u[1] = lo.y; f.u[2] = lo.z; f.u[3] = lo.w;
  f.u[4] = hi.x; f.u[5] = hi.y; f.u[6] = hi.z; f.u[7] = hi.w;
}

// Async global->LDS 16B copy (gfx1250 GLOBAL_LOAD_ASYNC_TO_LDS_B128, ASYNCcnt).
__device__ __forceinline__ void async_copy_b128(unsigned int lds_off,
                                                unsigned long long gaddr) {
  asm volatile("global_load_async_to_lds_b128 %0, %1, off"
               :: "v"(lds_off), "v"(gaddr) : "memory");
}
__device__ __forceinline__ void async_wait0() {
  asm volatile("s_wait_asynccnt 0x0" ::: "memory");
}

// ---------------------------------------------------------------------------
// Elementwise f32 -> bf16 conversion (vectorized x4).
// ---------------------------------------------------------------------------
__global__ __launch_bounds__(256) void cvt_f32_to_bf16(
    const float* __restrict__ in, __bf16* __restrict__ out)
{
  const size_t i = (size_t)blockIdx.x * 256 + threadIdx.x;
  const float4 f = ((const float4*)in)[i];
  union { __bf16 b[4]; uint2 u; } t;
  t.b[0] = (__bf16)f.x; t.b[1] = (__bf16)f.y;
  t.b[2] = (__bf16)f.z; t.b[3] = (__bf16)f.w;
  ((uint2*)out)[i] = t.u;
}

// ---------------------------------------------------------------------------
// Batched bf16 GEMM:  C = alpha * A @ W^T  (double-buffered async staging)
//   A: bf16 [M][K], W: bf16 [N][K].
// mode 0: Cout row-major f32 [M][N]
// mode 1: scatter to [B][H][S][D]; optional bf16 mirror Cbf.
// Block: 256 threads = 8 waves; tile 128(M) x 64(N); K chunk 32.
// ---------------------------------------------------------------------------
#define TM 128
#define TN 64
#define KC 32

__device__ __forceinline__ void stage_tiles(
    const __bf16* __restrict__ Ain, const __bf16* __restrict__ Win,
    int K, int tileM, int tileN, int kc, int tid,
    __bf16 (*lA)[KC], __bf16 (*lB)[KC])
{
  // A tile (TM x KC = 8 KB): 2 x 16B per thread.
  {
    const int b0 = tid * 16;            // bytes 0..4095
    const int r0 = b0 >> 6, c0 = b0 & 63;
    const __bf16* a0 = Ain + (size_t)(tileM + r0) * K + kc;
    async_copy_b128((unsigned int)(uintptr_t)&lA[r0][0] + c0,
                    (unsigned long long)(uintptr_t)a0 + c0);
    const int b1 = 4096 + tid * 16;     // bytes 4096..8191
    const int r1 = b1 >> 6, c1 = b1 & 63;
    const __bf16* a1 = Ain + (size_t)(tileM + r1) * K + kc;
    async_copy_b128((unsigned int)(uintptr_t)&lA[r1][0] + c1,
                    (unsigned long long)(uintptr_t)a1 + c1);
    if (kc + KC < K) __builtin_prefetch(a0 + KC, 0, 1);
  }
  // W tile (TN x KC = 4 KB): 1 x 16B per thread.
  {
    const int b0 = tid * 16;
    const int r0 = b0 >> 6, c0 = b0 & 63;
    const __bf16* w0 = Win + (size_t)(tileN + r0) * K + kc;
    async_copy_b128((unsigned int)(uintptr_t)&lB[r0][0] + c0,
                    (unsigned long long)(uintptr_t)w0 + c0);
  }
}

__global__ __launch_bounds__(256) void gemm_bf16(
    const __bf16* __restrict__ Ain, const __bf16* __restrict__ Win,
    float* __restrict__ Cout, __bf16* __restrict__ Cbf,
    int M, int N, int K, float alpha, int mode)
{
  __shared__ __align__(16) __bf16 lA[2][TM][KC];  // 2 x 8 KB ping-pong
  __shared__ __align__(16) __bf16 lB[2][TN][KC];  // 2 x 4 KB

  const int tid  = threadIdx.x;
  const int wave = tid >> 5;     // wave32
  const int lane = tid & 31;
  const int half = lane >> 4;
  const int l16  = lane & 15;
  const int tileM = blockIdx.y * TM;
  const int tileN = blockIdx.x * TN;

  v8f acc[4] = {};

  // Prologue: stage chunk 0 into buffer 0.
  stage_tiles(Ain, Win, K, tileM, tileN, 0, tid, lA[0], lB[0]);
  async_wait0();
  __syncthreads();

  int cur = 0;
  for (int kc = 0; kc < K; kc += KC) {
    const int nxt = cur ^ 1;
    // Overlap: DMA chunk kc+KC into the other buffer while computing kc.
    if (kc + KC < K)
      stage_tiles(Ain, Win, K, tileM, tileN, kc + KC, tid, lA[nxt], lB[nxt]);

    FragBF fa;
    load_frag(fa, &lA[cur][wave * 16 + l16][0], half);

    FragBF fb[4];
    #pragma unroll
    for (int nt = 0; nt < 4; ++nt)
      load_frag(fb[nt], &lB[cur][nt * 16 + l16][0], half);

    #pragma unroll
    for (int nt = 0; nt < 4; ++nt)
      acc[nt] = __builtin_amdgcn_wmma_f32_16x16x32_bf16(
          false, fa.v, false, fb[nt].v, (short)0, acc[nt], false, false);

    async_wait0();      // next chunk's DMA done (this wave)
    __syncthreads();    // all waves done computing cur + staging nxt
    cur = nxt;
  }

  // Epilogue. C/D layout: VGPR r -> M = r + 8*half, N = lane&15.
  #pragma unroll
  for (int nt = 0; nt < 4; ++nt) {
    #pragma unroll
    for (int r = 0; r < 8; ++r) {
      const int m = tileM + wave * 16 + r + 8 * half;
      const int n = tileN + nt * 16 + l16;
      const float val = alpha * acc[nt][r];
      if (mode == 0) {
        Cout[(size_t)m * N + n] = val;
      } else {
        const int b = m / S_DIM, s = m % S_DIM;
        const int h = n / D_DIM, d = n % D_DIM;
        const size_t o = (((size_t)b * H_DIM + h) * S_DIM + s) * D_DIM + d;
        if (Cout) Cout[o] = val;
        if (Cbf)  Cbf[o] = (__bf16)val;
      }
    }
  }
}

// ---------------------------------------------------------------------------
// Fused attention strip kernel. One block = one (b,h) x 16 query rows.
//   phase 1: scores = (Q K^T)/8 via WMMA, bf16 strip in LDS (16x2048 = 64 KB)
//   phase 2: softmax over the strip (half-wave shfl reductions),
//            normalized P streamed to global ctx (the tuple output)
//   phase 3: A_dot = P @ V via WMMA (per-wave K-partials, reduced via LDS)
// ---------------------------------------------------------------------------
#define ROWS 16

__global__ __launch_bounds__(256) void fused_attn(
    const __bf16* __restrict__ qg,   // [BH][S][D] bf16
    const __bf16* __restrict__ kg,   // [BH][S][D] bf16
    const __bf16* __restrict__ vTg,  // [BH][D][S] bf16
    float* __restrict__ ctx,         // [BH][S][S] f32 (tuple output)
    float* __restrict__ adot)        // [BH][S][D] f32
{
  __shared__ __align__(16) __bf16 strip[ROWS][S_DIM];   // 64 KB

  const int bh   = blockIdx.z;
  const int row0 = blockIdx.x * ROWS;
  const int tid  = threadIdx.x;
  const int wave = tid >> 5;
  const int lane = tid & 31;
  const int half = lane >> 4;
  const int l16  = lane & 15;

  const __bf16* qb = qg  + (size_t)bh * S_DIM * D_DIM;
  const __bf16* kb = kg  + (size_t)bh * S_DIM * D_DIM;
  const __bf16* vT = vTg + (size_t)bh * D_DIM * S_DIM;

  // Q fragments for the strip's 16 rows (K=64 -> 2 chunks).
  FragBF fq[2];
  {
    const int m = row0 + l16;
    #pragma unroll
    for (int c = 0; c < 2; ++c)
      load_frag(fq[c], qb + (size_t)m * D_DIM + c * 32, half);
  }

  // ---- phase 1: scores for this wave's 256-column key segment ----
  const int colBase = wave * 256;
  #pragma unroll 1
  for (int t = 0; t < 16; ++t) {
    const int ncol = colBase + t * 16 + l16;    // key position (B frag lane)
    v8f acc = {};
    #pragma unroll
    for (int c = 0; c < 2; ++c) {
      FragBF fk;
      load_frag(fk, kb + (size_t)ncol * D_DIM + c * 32, half);
      acc = __builtin_amdgcn_wmma_f32_16x16x32_bf16(
          false, fq[c].v, false, fk.v, (short)0, acc, false, false);
    }
    #pragma unroll
    for (int r = 0; r < 8; ++r)
      strip[r + 8 * half][colBase + t * 16 + l16] = (__bf16)(acc[r] * 0.125f);
  }
  __syncthreads();

  // ---- phase 2: softmax. Wave w owns rows 2w (lanes 0-15) / 2w+1 (16-31).
  const int srow = wave * 2 + half;
  float mx = -3.402823466e38f;
  for (int i = l16; i < S_DIM; i += 16)
    mx = fmaxf(mx, (float)strip[srow][i]);
  #pragma unroll
  for (int off = 8; off > 0; off >>= 1)
    mx = fmaxf(mx, __shfl_xor(mx, off, 16));

  float sum = 0.f;
  for (int i = l16; i < S_DIM; i += 16) {
    const float e = __expf((float)strip[srow][i] - mx);
    strip[srow][i] = (__bf16)e;
    sum += e;
  }
  #pragma unroll
  for (int off = 8; off > 0; off >>= 1)
    sum += __shfl_xor(sum, off, 16);
  const float inv = 1.f / sum;

  float* crow = ctx + ((size_t)bh * S_DIM + row0 + srow) * S_DIM;
  for (int i = l16; i < S_DIM; i += 16) {
    const float p = (float)strip[srow][i] * inv;
    strip[srow][i] = (__bf16)p;   // keep bf16 P for phase 3
    crow[i] = p;                  // stream tuple output (f32)
  }
  __syncthreads();

  // ---- phase 3: A_dot partials over this wave's K segment (256 keys) ----
  v8f pv[4] = {};
  #pragma unroll 1
  for (int c = 0; c < 8; ++c) {
    const int kc = colBase + c * 32;
    FragBF fp;                          // A = P rows from LDS (already bf16)
    load_frag(fp, &strip[l16][kc], half);

    FragBF fv[4];                       // B[k][n] = vT[n][k], contiguous in k
    #pragma unroll
    for (int nt = 0; nt < 4; ++nt)
      load_frag(fv[nt], vT + (size_t)(nt * 16 + l16) * S_DIM + kc, half);

    #pragma unroll
    for (int nt = 0; nt < 4; ++nt)
      pv[nt] = __builtin_amdgcn_wmma_f32_16x16x32_bf16(
          false, fp.v, false, fv[nt].v, (short)0, pv[nt], false, false);
  }
  __syncthreads();   // strip reads complete before re-use as f32 scratch

  // Cross-wave reduction through the strip memory (8 x 16 x 64 f32 = 32 KB).
  float* part = reinterpret_cast<float*>(&strip[0][0]);
  #pragma unroll
  for (int nt = 0; nt < 4; ++nt) {
    #pragma unroll
    for (int r = 0; r < 8; ++r) {
      const int mm = r + 8 * half;
      const int nn = nt * 16 + l16;
      part[(size_t)wave * 1024 + mm * 64 + nn] = pv[nt][r];
    }
  }
  __syncthreads();

  for (int i = tid; i < ROWS * D_DIM; i += 256) {
    float s = 0.f;
    #pragma unroll
    for (int w = 0; w < 8; ++w) s += part[(size_t)w * 1024 + i];
    const int mm = i / D_DIM, nn = i % D_DIM;
    adot[((size_t)bh * S_DIM + row0 + mm) * D_DIM + nn] = s;
  }
}

// ---------------------------------------------------------------------------
// Linear-attention memory read: A_mem = (elu(q)+1) @ M_h / (elu(q)+1)·z_h.
// ---------------------------------------------------------------------------
__global__ __launch_bounds__(64) void mem_read_kernel(
    const float* __restrict__ q, const float* __restrict__ memory,
    const float* __restrict__ z, float* __restrict__ amem)
{
  __shared__ float eq[D_DIM];
  const int gr = blockIdx.x;                 // 0 .. B*H*S-1
  const int h  = (gr / S_DIM) % H_DIM;
  const int d  = threadIdx.x;

  const float qa = q[(size_t)gr * D_DIM + d];
  eq[d] = (qa > 0.f) ? (qa + 1.f) : __expf(qa);   // elu(q)+1
  __syncthreads();

  float den = 0.f, num = 0.f;
  #pragma unroll 8
  for (int k = 0; k < D_DIM; ++k) {
    const float e = eq[k];
    den += e * z[h * D_DIM + k];
    num += e * memory[((size_t)h * D_DIM + k) * D_DIM + d];
  }
  amem[(size_t)gr * D_DIM + d] = num / den;
}

// ---------------------------------------------------------------------------
// V transpose per (b,h): f32 [S][D] -> bf16 [D][S].
// ---------------------------------------------------------------------------
__global__ __launch_bounds__(256) void transpose_SD_to_DS(
    const float* __restrict__ src, __bf16* __restrict__ dst)
{
  const size_t bh = blockIdx.z;
  const int idx = blockIdx.x * 256 + threadIdx.x;   // over S*D
  const int s = idx / D_DIM, d = idx % D_DIM;
  dst[(bh * D_DIM + d) * S_DIM + s] =
      (__bf16)src[(bh * S_DIM + s) * D_DIM + d];
}

// ---------------------------------------------------------------------------
// Gated combine -> bf16 xpre[b,s,h*D+d] = g*A_mem + (1-g)*A_dot.
// ---------------------------------------------------------------------------
__global__ __launch_bounds__(256) void combine_kernel(
    const float* __restrict__ amem, const float* __restrict__ adot,
    const float* __restrict__ beta, __bf16* __restrict__ xpre)
{
  const size_t i = (size_t)blockIdx.x * 256 + threadIdx.x;  // over B*H*S*D
  const int d = (int)(i % D_DIM);
  const int s = (int)((i / D_DIM) % S_DIM);
  const int h = (int)((i / ((size_t)D_DIM * S_DIM)) % H_DIM);
  const int b = (int)(i / ((size_t)D_DIM * S_DIM * H_DIM));
  const float g = 1.f / (1.f + __expf(-beta[h]));
  const float val = g * amem[i] + (1.f - g) * adot[i];
  xpre[(((size_t)b * S_DIM + s) * H_DIM + h) * D_DIM + d] = (__bf16)val;
}

// ---------------------------------------------------------------------------
extern "C" void kernel_launch(void* const* d_in, const int* in_sizes, int n_in,
                              void* d_out, int out_size, void* d_ws, size_t ws_size,
                              hipStream_t stream) {
  (void)in_sizes; (void)n_in; (void)out_size; (void)ws_size;

  const float* query  = (const float*)d_in[0];
  const float* key_   = (const float*)d_in[1];
  const float* value  = (const float*)d_in[2];
  const float* Wq     = (const float*)d_in[3];
  const float* Wk     = (const float*)d_in[4];
  const float* Wv     = (const float*)d_in[5];
  const float* Wo     = (const float*)d_in[6];
  const float* beta   = (const float*)d_in[7];
  const float* memory = (const float*)d_in[8];
  const float* zvec   = (const float*)d_in[9];

  float* out_x = (float*)d_out;                              // [B,S,E]
  float* ctx   = out_x + (size_t)B_DIM * S_DIM * E_DIM;      // [B,H,S,S]

  const int    M    = B_DIM * S_DIM;                 // 4096
  const int    BH   = B_DIM * H_DIM;                 // 32
  const size_t SLOT = (size_t)M * HD_DIM;            // 4M elements
  const size_t WSZ  = (size_t)HD_DIM * E_DIM;        // 1M elements

  // f32 workspace region: q, v, amem, adot (4 x 16 MB)
  float* ws    = (float*)d_ws;
  float* qf    = ws + 0 * SLOT;
  float* vf    = ws + 1 * SLOT;
  float* amem  = ws + 2 * SLOT;
  float* adot  = ws + 3 * SLOT;
  // bf16 region after that
  __bf16* bws   = (__bf16*)(ws + 4 * SLOT);
  __bf16* qin16 = bws + 0 * SLOT;    // bf16 query [M][E]
  __bf16* kin16 = bws + 1 * SLOT;
  __bf16* vin16 = bws + 2 * SLOT;
  __bf16* q16   = bws + 3 * SLOT;    // head-split bf16 [BH][S][D]
  __bf16* k16   = bws + 4 * SLOT;
  __bf16* vT16  = bws + 5 * SLOT;    // [BH][D][S]
  __bf16* xpre16= bws + 6 * SLOT;    // [M][HD]
  __bf16* wq16  = bws + 7 * SLOT + 0 * WSZ;
  __bf16* wk16  = bws + 7 * SLOT + 1 * WSZ;
  __bf16* wv16  = bws + 7 * SLOT + 2 * WSZ;
  __bf16* wo16  = bws + 7 * SLOT + 3 * WSZ;

  dim3 blk(256);

  // 0) Convert GEMM operands to bf16 (one pass each).
  cvt_f32_to_bf16<<<dim3((unsigned)(SLOT / 1024)), blk, 0, stream>>>(query, qin16);
  cvt_f32_to_bf16<<<dim3((unsigned)(SLOT / 1024)), blk, 0, stream>>>(key_,  kin16);
  cvt_f32_to_bf16<<<dim3((unsigned)(SLOT / 1024)), blk, 0, stream>>>(value, vin16);
  cvt_f32_to_bf16<<<dim3((unsigned)(WSZ / 1024)),  blk, 0, stream>>>(Wq, wq16);
  cvt_f32_to_bf16<<<dim3((unsigned)(WSZ / 1024)),  blk, 0, stream>>>(Wk, wk16);
  cvt_f32_to_bf16<<<dim3((unsigned)(WSZ / 1024)),  blk, 0, stream>>>(Wv, wv16);
  cvt_f32_to_bf16<<<dim3((unsigned)(WSZ / 1024)),  blk, 0, stream>>>(Wo, wo16);

  // 1) Projections (head-split scatter epilogue; f32 and/or bf16 mirrors).
  gemm_bf16<<<dim3(HD_DIM / TN, M / TM, 1), blk, 0, stream>>>(
      qin16, wq16, qf, q16, M, HD_DIM, E_DIM, 1.f, 1);       // q: f32 + bf16
  gemm_bf16<<<dim3(HD_DIM / TN, M / TM, 1), blk, 0, stream>>>(
      kin16, wk16, nullptr, k16, M, HD_DIM, E_DIM, 1.f, 1);  // k: bf16 only
  gemm_bf16<<<dim3(HD_DIM / TN, M / TM, 1), blk, 0, stream>>>(
      vin16, wv16, vf, nullptr, M, HD_DIM, E_DIM, 1.f, 1);   // v: f32 only

  // 2) V^T per (b,h), f32 -> bf16.
  transpose_SD_to_DS<<<dim3((S_DIM * D_DIM) / 256, 1, BH), blk, 0, stream>>>(vf, vT16);

  // 3) Fused scores + softmax + P-write + P@V.
  fused_attn<<<dim3(S_DIM / ROWS, 1, BH), blk, 0, stream>>>(q16, k16, vT16, ctx, adot);

  // 4) Linear-attention memory read.
  mem_read_kernel<<<dim3(BH * S_DIM), dim3(64), 0, stream>>>(qf, memory, zvec, amem);

  // 5) Gated combine into bf16 [B,S,H*D].
  combine_kernel<<<dim3((unsigned)(SLOT / 256)), blk, 0, stream>>>(
      amem, adot, beta, xpre16);

  // 6) Output projection: x = xpre @ Wo^T (f32 out).
  gemm_bf16<<<dim3(E_DIM / TN, M / TM, 1), blk, 0, stream>>>(
      xpre16, wo16, out_x, nullptr, M, E_DIM, HD_DIM, 1.f, 0);
}
